// MultiHeadAttention_36404142801403
// MI455X (gfx1250) — matrix operand order repaired and checked by reading, not compiled, not run on hardware
//
#include <hip/hip_runtime.h>

#define D_MODEL 1024
#define N_HEADS 16
#define HEAD_DIM 64
#define BATCH 4
#define SEQ 2048
#define M_ROWS (BATCH * SEQ)   // 8192

typedef __attribute__((ext_vector_type(16))) __bf16 bf16x16;
typedef __attribute__((ext_vector_type(8)))  float  f32x8;
typedef __attribute__((ext_vector_type(4)))  unsigned int u32x4;

union BF16Frag { bf16x16 v; u32x4 q[2]; };

__device__ __forceinline__ unsigned short f2bf(float f) {
  unsigned int u = __float_as_uint(f);
  u += 0x7fffu + ((u >> 16) & 1u);        // round-to-nearest-even
  return (unsigned short)(u >> 16);
}

// wave32 xor-shuffle via ds_swizzle (group-of-32: and=0x1f, or=0, xor=MASK)
template <int MASK>
__device__ __forceinline__ float swz_xor(float v) {
  return __int_as_float(__builtin_amdgcn_ds_swizzle(__float_as_int(v), (MASK << 10) | 0x1f));
}
__device__ __forceinline__ float rmax16(float v) {
  v = fmaxf(v, swz_xor<1>(v));
  v = fmaxf(v, swz_xor<2>(v));
  v = fmaxf(v, swz_xor<4>(v));
  v = fmaxf(v, swz_xor<8>(v));
  return v;
}
__device__ __forceinline__ float radd16(float v) {
  v += swz_xor<1>(v);
  v += swz_xor<2>(v);
  v += swz_xor<4>(v);
  v += swz_xor<8>(v);
  return v;
}

__global__ __launch_bounds__(256) void f32_to_bf16_kernel(
    const float* __restrict__ src, unsigned short* __restrict__ dst, int n) {
  int i = blockIdx.x * blockDim.x + threadIdx.x;
  int stride = gridDim.x * blockDim.x;
  for (; i < n; i += stride) dst[i] = f2bf(src[i]);
}

// Y[M,N] = X[M,K] * W[N,K]^T   (torch-style linear), bf16 in, f32 accumulate.
// MODE 0: store bf16 row-major [M,N]
// MODE 1: store bf16 transposed per head -> Vt[B][H][Hd][S]
// MODE 2: store f32 row-major [M,N]
template <int MODE>
__global__ __launch_bounds__(256) void gemm_xwT_kernel(
    const unsigned short* __restrict__ X,
    const unsigned short* __restrict__ W,
    unsigned short* __restrict__ Yb,
    float* __restrict__ Yf,
    int M, int N, int K) {
  int wave = blockIdx.x * 8 + (threadIdx.x >> 5);
  int nTiles = N >> 6;                       // 64-col tiles
  int mTile = wave / nTiles;
  int nTile = wave - mTile * nTiles;
  if ((mTile << 4) >= M) return;             // wave-uniform guard
  int lane = threadIdx.x & 31;
  int l15 = lane & 15, hi = lane >> 4;
  int row0 = mTile << 4, col0 = nTile << 6;

  f32x8 acc[4];
#pragma unroll
  for (int c = 0; c < 4; ++c)
#pragma unroll
    for (int r = 0; r < 8; ++r) acc[c][r] = 0.f;

  // A frag: lane holds row (row0+l15); chunks at k + hi*8 and k + 16 + hi*8
  const unsigned short* aPtr = X + (size_t)(row0 + l15) * K + hi * 8;
  // B frag: lane holds W-row (col0+c*16+l15); chunks at k + hi*8 and k + 16 + hi*8
  const unsigned short* bPtr = W + (size_t)(col0 + l15) * K + hi * 8;

  for (int k = 0; k < K; k += 32) {
    BF16Frag a;
    a.q[0] = *(const u32x4*)(aPtr + k);
    a.q[1] = *(const u32x4*)(aPtr + k + 16);
#pragma unroll
    for (int c = 0; c < 4; ++c) {
      const unsigned short* bp = bPtr + (size_t)c * 16 * K + k;
      BF16Frag bfr;
      bfr.q[0] = *(const u32x4*)(bp);
      bfr.q[1] = *(const u32x4*)(bp + 16);
      acc[c] = __builtin_amdgcn_wmma_f32_16x16x32_bf16(
          false, a.v, false, bfr.v, (short)0, acc[c], false, false);
    }
  }

#pragma unroll
  for (int c = 0; c < 4; ++c) {
#pragma unroll
    for (int r = 0; r < 8; ++r) {
      int row = row0 + r + hi * 8;           // C layout: reg r -> M = r + 8*hi
      int col = col0 + c * 16 + l15;         // lane%16 -> N
      float val = acc[c][r];
      if (MODE == 0) {
        Yb[(size_t)row * N + col] = f2bf(val);
      } else if (MODE == 1) {
        int bb = row >> 11;                  // row / SEQ
        int ss = row & (SEQ - 1);
        int hh = col >> 6;                   // col / HEAD_DIM
        int hd = col & (HEAD_DIM - 1);
        Yb[((size_t)(bb * N_HEADS + hh) * HEAD_DIM + hd) * SEQ + ss] = f2bf(val);
      } else {
        Yf[(size_t)row * N + col] = val;
      }
    }
  }
}

// One wave per (batch, head, 16-query-row tile). Streaming (flash) softmax.
__global__ __launch_bounds__(256) void flash_attn_kernel(
    const unsigned short* __restrict__ Q,    // [B*S, D] bf16
    const unsigned short* __restrict__ Kc,   // [B*S, D] bf16
    const unsigned short* __restrict__ Vt,   // [B,H,Hd,S] bf16
    const int* __restrict__ amask,           // [B,S]
    unsigned short* __restrict__ Outb) {     // [B*S, D] bf16
  __shared__ __align__(16) unsigned short pTile[8][16 * 32];  // 1KB per wave
  int wvid = threadIdx.x >> 5;
  int wave = blockIdx.x * 8 + wvid;
  const int qTiles = SEQ / 16;
  int bh = wave / qTiles;
  int qi = wave - bh * qTiles;
  int b = bh >> 4;                           // / N_HEADS
  int h = bh & (N_HEADS - 1);
  int lane = threadIdx.x & 31;
  int l15 = lane & 15, hi = lane >> 4;
  int q0 = qi * 16;

  // Q A-fragments: K = 64 -> two 16x32 chunks
  const unsigned short* qBase =
      Q + (size_t)(b * SEQ + q0 + l15) * D_MODEL + h * HEAD_DIM + hi * 8;
  BF16Frag aq0, aq1;
  aq0.q[0] = *(const u32x4*)(qBase);
  aq0.q[1] = *(const u32x4*)(qBase + 16);
  aq1.q[0] = *(const u32x4*)(qBase + 32);
  aq1.q[1] = *(const u32x4*)(qBase + 48);

  f32x8 O[4];
#pragma unroll
  for (int c = 0; c < 4; ++c)
#pragma unroll
    for (int r = 0; r < 8; ++r) O[c][r] = 0.f;

  float mrow[8], lrow[8];
#pragma unroll
  for (int r = 0; r < 8; ++r) { mrow[r] = -3.0e38f; lrow[r] = 0.f; }

  const float scale = 0.125f;                // 1/sqrt(64)
  const int kEnd = q0 + 16;                  // causal: max key + 1

  for (int j = 0; j < kEnd; j += 32) {
    // ---- scores: S = Q * K^T for 32 keys (two 16-key N-chunks) ----
    f32x8 sc[2];
#pragma unroll
    for (int c = 0; c < 2; ++c) {
#pragma unroll
      for (int r = 0; r < 8; ++r) sc[c][r] = 0.f;
      const unsigned short* kp =
          Kc + (size_t)(b * SEQ + j + c * 16 + l15) * D_MODEL + h * HEAD_DIM + hi * 8;
      BF16Frag bk0, bk1;
      bk0.q[0] = *(const u32x4*)(kp);
      bk0.q[1] = *(const u32x4*)(kp + 16);
      bk1.q[0] = *(const u32x4*)(kp + 32);
      bk1.q[1] = *(const u32x4*)(kp + 48);
      sc[c] = __builtin_amdgcn_wmma_f32_16x16x32_bf16(
          false, aq0.v, false, bk0.v, (short)0, sc[c], false, false);
      sc[c] = __builtin_amdgcn_wmma_f32_16x16x32_bf16(
          false, aq1.v, false, bk1.v, (short)0, sc[c], false, false);
    }

    // ---- mask + online softmax (rows live in 16-lane groups) ----
    int kg0 = j + l15;
    int kg1 = j + 16 + l15;
    int km0 = amask[b * SEQ + kg0] != 0;
    int km1 = amask[b * SEQ + kg1] != 0;

    float alpha[8], p0[8], p1[8];
#pragma unroll
    for (int r = 0; r < 8; ++r) {
      int mg = q0 + r + hi * 8;
      float s0 = (km0 && kg0 <= mg) ? sc[0][r] * scale : -3.0e38f;
      float s1 = (km1 && kg1 <= mg) ? sc[1][r] * scale : -3.0e38f;
      float mx = rmax16(fmaxf(s0, s1));
      mx = fmaxf(mx, mrow[r]);
      float a = __expf(mrow[r] - mx);
      p0[r] = (s0 > -1.0e30f) ? __expf(s0 - mx) : 0.f;
      p1[r] = (s1 > -1.0e30f) ? __expf(s1 - mx) : 0.f;
      float rs = radd16(p0[r] + p1[r]);
      lrow[r] = lrow[r] * a + rs;
      mrow[r] = mx;
      alpha[r] = a;
    }
#pragma unroll
    for (int c = 0; c < 4; ++c)
#pragma unroll
      for (int r = 0; r < 8; ++r) O[c][r] *= alpha[r];

    // ---- re-swizzle P: C layout -> LDS -> A layout (16x32 bf16) ----
    unsigned short* pt = pTile[wvid];
#pragma unroll
    for (int r = 0; r < 8; ++r) {
      pt[(r + hi * 8) * 32 + l15]      = f2bf(p0[r]);
      pt[(r + hi * 8) * 32 + 16 + l15] = f2bf(p1[r]);
    }
    asm volatile("s_wait_dscnt 0" ::: "memory");
    BF16Frag pa;
    const unsigned short* pp = pt + l15 * 32 + hi * 8;
    pa.q[0] = *(const u32x4*)(pp);
    pa.q[1] = *(const u32x4*)(pp + 16);

    // ---- O += P * V  (V pre-transposed: contiguous along keys) ----
#pragma unroll
    for (int c = 0; c < 4; ++c) {
      const unsigned short* vp =
          Vt + (size_t)((b * N_HEADS + h) * HEAD_DIM + c * 16 + l15) * SEQ + j + hi * 8;
      BF16Frag bv;
      bv.q[0] = *(const u32x4*)(vp);
      bv.q[1] = *(const u32x4*)(vp + 16);
      O[c] = __builtin_amdgcn_wmma_f32_16x16x32_bf16(
          false, pa.v, false, bv.v, (short)0, O[c], false, false);
    }
  }

  // ---- normalize and store attention output (bf16) ----
#pragma unroll
  for (int c = 0; c < 4; ++c)
#pragma unroll
    for (int r = 0; r < 8; ++r) {
      int row = q0 + r + hi * 8;
      int col = h * HEAD_DIM + c * 16 + l15;
      Outb[(size_t)(b * SEQ + row) * D_MODEL + col] = f2bf(O[c][r] / lrow[r]);
    }
}

extern "C" void kernel_launch(void* const* d_in, const int* in_sizes, int n_in,
                              void* d_out, int out_size, void* d_ws, size_t ws_size,
                              hipStream_t stream) {
  const float* x  = (const float*)d_in[0];
  const int*   am = (const int*)d_in[1];
  const float* wq = (const float*)d_in[2];
  const float* wk = (const float*)d_in[3];
  const float* wv = (const float*)d_in[4];
  const float* wo = (const float*)d_in[5];
  float* out = (float*)d_out;

  unsigned char* p = (unsigned char*)d_ws;
  const size_t xBytes = (size_t)M_ROWS * D_MODEL * sizeof(unsigned short);  // 16 MB
  const size_t wBytes = (size_t)D_MODEL * D_MODEL * sizeof(unsigned short); // 2 MB
  unsigned short* xbf = (unsigned short*)p; p += xBytes;
  unsigned short* wqb = (unsigned short*)p; p += wBytes;
  unsigned short* wkb = (unsigned short*)p; p += wBytes;
  unsigned short* wvb = (unsigned short*)p; p += wBytes;
  unsigned short* wob = (unsigned short*)p; p += wBytes;
  unsigned short* Qb  = (unsigned short*)p; p += xBytes;
  unsigned short* Kb  = (unsigned short*)p; p += xBytes;
  unsigned short* Vt  = (unsigned short*)p; p += xBytes;
  unsigned short* Ab  = (unsigned short*)p; p += xBytes;   // total ~88 MB

  f32_to_bf16_kernel<<<2048, 256, 0, stream>>>(x,  xbf, M_ROWS * D_MODEL);
  f32_to_bf16_kernel<<<512,  256, 0, stream>>>(wq, wqb, D_MODEL * D_MODEL);
  f32_to_bf16_kernel<<<512,  256, 0, stream>>>(wk, wkb, D_MODEL * D_MODEL);
  f32_to_bf16_kernel<<<512,  256, 0, stream>>>(wv, wvb, D_MODEL * D_MODEL);
  f32_to_bf16_kernel<<<512,  256, 0, stream>>>(wo, wob, D_MODEL * D_MODEL);

  const int gemmBlocks = (M_ROWS / 16) * (D_MODEL / 64) / 8;   // 1024 blocks x 8 waves
  gemm_xwT_kernel<0><<<gemmBlocks, 256, 0, stream>>>(xbf, wqb, Qb, nullptr,
                                                     M_ROWS, D_MODEL, D_MODEL);
  gemm_xwT_kernel<0><<<gemmBlocks, 256, 0, stream>>>(xbf, wkb, Kb, nullptr,
                                                     M_ROWS, D_MODEL, D_MODEL);
  gemm_xwT_kernel<1><<<gemmBlocks, 256, 0, stream>>>(xbf, wvb, Vt, nullptr,
                                                     M_ROWS, D_MODEL, D_MODEL);

  const int flashBlocks = BATCH * N_HEADS * (SEQ / 16) / 8;    // 1024
  flash_attn_kernel<<<flashBlocks, 256, 0, stream>>>(Qb, Kb, Vt, am, Ab);

  gemm_xwT_kernel<2><<<gemmBlocks, 256, 0, stream>>>(Ab, wob, nullptr, out,
                                                     M_ROWS, D_MODEL, D_MODEL);

  (void)in_sizes; (void)n_in; (void)out_size; (void)ws_size;
}
